// ConvVQ_19310172963583
// MI455X (gfx1250) — matrix operand order repaired and checked by reading, not compiled, not run on hardware
//
#include <hip/hip_runtime.h>

typedef __attribute__((ext_vector_type(16))) _Float16 v16h;
typedef __attribute__((ext_vector_type(8)))  _Float16 v8h;
typedef __attribute__((ext_vector_type(2)))  _Float16 v2h;
typedef __attribute__((ext_vector_type(8)))  float    v8f;

#define Bc   64
#define Dc   128
#define Hc   32
#define Wc   32
#define Kc   1024
#define HWc  1024            // H*W
#define NPOS 65536           // B*H*W
#define APAD 136             // 128 + 8 f16 pad -> conflict-free LDS banks

// ---------------------------------------------------------------------------
// Kernel 1: per-embedding squared norms + f32 -> f16 embedding table.
// ---------------------------------------------------------------------------
__global__ __launch_bounds__(256) void vq_prep(const float* __restrict__ emb,
                                               float* __restrict__ sums,
                                               _Float16* __restrict__ embh) {
  int k = blockIdx.x * 256 + threadIdx.x;
  if (k >= Kc) return;
  const float* row  = emb  + (size_t)k * Dc;
  _Float16*    hrow = embh + (size_t)k * Dc;
  float s = 0.f;
#pragma unroll
  for (int d = 0; d < Dc; d += 4) {
    float4 v = *(const float4*)(row + d);
    s = fmaf(v.x, v.x, s); s = fmaf(v.y, v.y, s);
    s = fmaf(v.z, v.z, s); s = fmaf(v.w, v.w, s);
    v2h p0; p0[0] = (_Float16)v.x; p0[1] = (_Float16)v.y;
    v2h p1; p1[0] = (_Float16)v.z; p1[1] = (_Float16)v.w;
    *(v2h*)(hrow + d)     = p0;
    *(v2h*)(hrow + d + 2) = p1;
  }
  sums[k] = s;
}

// ---------------------------------------------------------------------------
// Kernel 2: fused distance-GEMM (WMMA f16) + online argmin + scatter.
// One wave32 per 16-position tile; 4 waves per block. Double-buffered B.
// ---------------------------------------------------------------------------
__global__ __launch_bounds__(128) void vq_main(const float* __restrict__ z_e,
                                               const float* __restrict__ embf,
                                               const float* __restrict__ sums,
                                               const _Float16* __restrict__ embh,
                                               float* __restrict__ out,
                                               float* __restrict__ zq) {
  __shared__ __align__(16) _Float16 zA[4][16 * APAD];
  __shared__ __align__(16) int idxS[4][16];

  const int w    = threadIdx.x >> 5;   // wave within block
  const int lane = threadIdx.x & 31;
  const int li   = lane & 15;          // row (A/C) or column (B) id
  const int hf   = lane >> 4;          // half-wave select

  const int tile = blockIdx.x * 4 + w;
  const int p0   = tile * 16;
  const int b    = p0 >> 10;           // image index (16 | 1024, no crossing)
  const int hw0  = p0 & (HWc - 1);
  const float* zbase = z_e + (size_t)b * Dc * HWc + hw0;

  _Float16* zw = zA[w];

  // ---- Stage z tile (16 pos x 128 dim) as f16 in LDS, row-major per pos ----
  {
    const int doff = hf << 1;          // lanes 0-15: dims d0,d0+1; lanes 16-31: d0+2,d0+3
    for (int d0 = 0; d0 < Dc; d0 += 4) {
      int dd = d0 + doff;
      float x0 = zbase[dd * HWc + li];         // 64B-coalesced per 16-lane group
      float x1 = zbase[(dd + 1) * HWc + li];
      v2h p; p[0] = (_Float16)x0; p[1] = (_Float16)x1;
      *(v2h*)(zw + li * APAD + dd) = p;        // conflict-free (4*li + dd/2 banks)
    }
  }

  // ---- A fragments: loop-invariant, pulled into VGPRs once ----
  const _Float16* arow = zw + li * APAD + hf * 16;
  v16h af[4];
#pragma unroll
  for (int s = 0; s < 4; ++s) {
    v8h a0 = *(const v8h*)(arow + 32 * s);        // ds_load_b128
    v8h a1 = *(const v8h*)(arow + 32 * s + 8);
    af[s] = __builtin_shufflevector(a0, a1, 0,1,2,3,4,5,6,7,8,9,10,11,12,13,14,15);
  }

  float bestV[8]; int bestI[8];
#pragma unroll
  for (int r = 0; r < 8; ++r) { bestV[r] = 3.4e38f; bestI[r] = 0; }

  // B fragment source: column li = embedding row, K-half by hf (contiguous k = 16*hf + e per 32-chunk, same as A).
  const _Float16* bbase = embh + (size_t)li * Dc + hf * 16;

  auto loadB = [&](int kt, v8h* bf, float& sn) {
    const _Float16* brow = bbase + (size_t)kt * 16 * Dc;
#pragma unroll
    for (int s = 0; s < 4; ++s) {
      bf[2 * s]     = *(const v8h*)(brow + 32 * s);      // global_load_b128 (L2 hot)
      bf[2 * s + 1] = *(const v8h*)(brow + 32 * s + 8);
    }
    sn = sums[kt * 16 + li];
  };

  auto step = [&](const v8h* bf, float sn, int j) {
    v8f c = {};
#pragma unroll
    for (int s = 0; s < 4; ++s) {
      v16h bm = __builtin_shufflevector(bf[2 * s], bf[2 * s + 1],
                                        0,1,2,3,4,5,6,7,8,9,10,11,12,13,14,15);
      c = __builtin_amdgcn_wmma_f32_16x16x32_f16(false, af[s], false, bm,
                                                 (short)0, c, false, false);
    }
    // Per-lane j strictly increases with kt, so strict '<' keeps first occurrence.
#pragma unroll
    for (int r = 0; r < 8; ++r) {
      float v = fmaf(-2.f, c[r], sn);            // ||e||^2 - 2 z.e
      if (v < bestV[r]) { bestV[r] = v; bestI[r] = j; }
    }
  };

  // ---- Ping-pong double-buffered main loop over 64 tiles of 16 embeddings ----
  v8h b0[8], b1[8];
  float s0, s1;
  loadB(0, b0, s0);
  loadB(1, b1, s1);
  for (int kt = 0; kt < 64; kt += 2) {
    step(b0, s0, kt * 16 + li);
    loadB((kt + 2) & 63, b0, s0);                // prefetch; wraps harmlessly at end
    step(b1, s1, (kt + 1) * 16 + li);
    loadB((kt + 3) & 63, b1, s1);
  }

  // ---- Butterfly argmin across the 16 lanes sharing each row group ----
#pragma unroll
  for (int m = 1; m < 16; m <<= 1) {
#pragma unroll
    for (int r = 0; r < 8; ++r) {
      float ov = __shfl_xor(bestV[r], m, 32);
      int   oi = __shfl_xor(bestI[r], m, 32);
      if ((ov < bestV[r]) || ((ov == bestV[r]) && (oi < bestI[r]))) {
        bestV[r] = ov; bestI[r] = oi;
      }
    }
  }
  if (li == 0) {
#pragma unroll
    for (int r = 0; r < 8; ++r) idxS[w][hf * 8 + r] = bestI[r];   // rows 8*hf+r
  }

  // ---- Writeback: z_q = emb[idx], out = (z_q - z) + z  (reference rounding) ----
  // Each lane owns 4 consecutive positions at one dim-plane -> float4 traffic.
  const int i0 = (lane & 3) << 2;      // positions i0..i0+3
  const int dq = lane >> 2;            // dim offset 0..7
  int4 idx4 = *(const int4*)&idxS[w][i0];        // same-wave LDS, in-order
  const float* ep0 = embf + (size_t)idx4.x * Dc;
  const float* ep1 = embf + (size_t)idx4.y * Dc;
  const float* ep2 = embf + (size_t)idx4.z * Dc;
  const float* ep3 = embf + (size_t)idx4.w * Dc;
  float* outp = out + (size_t)b * Dc * HWc + hw0;
  float* zqp  = zq  + (size_t)b * Dc * HWc + hw0;
#pragma unroll 4
  for (int t = 0; t < 16; ++t) {
    int d = t * 8 + dq;
    float4 zv = *(const float4*)(zbase + d * HWc + i0);      // b128, coalesced
    float4 ev = make_float4(ep0[d], ep1[d], ep2[d], ep3[d]); // gather, L2 hot
    float4 ov;
    ov.x = (ev.x - zv.x) + zv.x;
    ov.y = (ev.y - zv.y) + zv.y;
    ov.z = (ev.z - zv.z) + zv.z;
    ov.w = (ev.w - zv.w) + zv.w;
    *(float4*)(zqp  + d * HWc + i0) = ev;                    // b128 stores,
    *(float4*)(outp + d * HWc + i0) = ov;                    // 64B per 4-lane group
  }
}

// ---------------------------------------------------------------------------
extern "C" void kernel_launch(void* const* d_in, const int* in_sizes, int n_in,
                              void* d_out, int out_size, void* d_ws, size_t ws_size,
                              hipStream_t stream) {
  const float* z_e = (const float*)d_in[0];     // (B, D, H, W) f32
  const float* emb = (const float*)d_in[1];     // (K, D) f32

  float*     sums = (float*)d_ws;                       // K floats
  _Float16*  embh = (_Float16*)((char*)d_ws + 4096);    // K*D f16 table

  float* out = (float*)d_out;                   // first output
  float* zq  = out + (size_t)Bc * Dc * Hc * Wc; // second output (concatenated)

  vq_prep<<<(Kc + 255) / 256, 256, 0, stream>>>(emb, sums, embh);
  vq_main<<<NPOS / 16 / 4, 128, 0, stream>>>(z_e, emb, sums, embh, out, zq);
}
